// MyRuleNetwork_876173328583
// MI455X (gfx1250) — matrix operand order repaired and checked by reading, not compiled, not run on hardware
//
#include <hip/hip_runtime.h>
#include <stdint.h>

typedef _Float16 half_t;
typedef __attribute__((ext_vector_type(16))) _Float16 v16h;
typedef __attribute__((ext_vector_type(8)))  float    v8f;

union FragH { v16h v; uint32_t u[8]; };
union FragF { v8f  v; float    f[8]; };

__device__ __forceinline__ v8f zero8() {
    FragF z;
#pragma unroll
    for (int j = 0; j < 8; ++j) z.f[j] = 0.0f;
    return z.v;
}

__device__ __forceinline__ uint32_t pack_h2(float lo, float hi) {
    union { _Float16 h[2]; uint32_t u; } t;
    t.h[0] = (_Float16)lo; t.h[1] = (_Float16)hi;
    return t.u;
}

// ---------------------------------------------------------------------------
// Problem constants (from reference)
#define BB        65536
#define VARD      100
#define HIDD      200     // rule_mlp input dim (pad to 224 = 7 k-steps)
#define MLPH      128     // hidden width (4 k-steps for GEMM2)
#define NRULES    4
#define SCORK     300     // hidden flat dim (pad to 320 = 10 k-steps)

// Main-kernel tiling
#define BATCH_TILE 128
#define NWAVES     8
#define HSTRIDE    324    // f16 elems per staged hidden row (>=320, bank-friendly)
#define MIDSTRIDE  132    // f16 elems per mid row (>=128, bank-friendly)

// Workspace offsets (bytes)
#define WS_S      0                 // S[8][300] f32          : 9600 B
#define WS_SFRAG  10240             // 10*8*32 dwords         : 10240 B
#define WS_W1F    20480             // 4*7*8*8*32 dwords      : 229376 B
#define WS_W2F    249856            // 4*4*7*8*32 dwords      : 114688 B
// total ~364.5 KB

// ---------------------------------------------------------------------------
// Prep A: fold SelectAttention into S[8][300]:  scores[b,p] = hidden[b,:] . S[p,:]
// p = r*2 + v (matches scores.reshape(b,-1) flatten order); 1/sqrt(32) folded in.
__global__ __launch_bounds__(256) void prep_scores(
    const float* __restrict__ rule_emb,  // (4,64)
    const float* __restrict__ Wq,        // (64,32)
    const float* __restrict__ bq,        // (32)
    const float* __restrict__ Wk,        // (2,200,32)
    float* __restrict__ S)               // (8,300)
{
    __shared__ float s_read[NRULES * 32];
    const int t = threadIdx.x;
    if (t < NRULES * 32) {
        const int r = t >> 5, k = t & 31;
        float acc = bq[k];
        for (int d = 0; d < 64; ++d)
            acc += rule_emb[r * 64 + d] * Wq[d * 32 + k];
        s_read[t] = acc;
    }
    __syncthreads();
    const float scale = 0.17677669529663687f;  // 1/sqrt(32)
    for (int j = t; j < 8 * SCORK; j += 256) {
        const int p = j / SCORK, d = j - p * SCORK;
        const int r = p >> 1, v = p & 1;
        int dd = -1;
        if (d < 100)        { if (v == 0) dd = d; }
        else if (d < 200)   { if (v == 1) dd = d - 100; }
        else                { dd = 100 + (d - 200); }  // op slot feeds both vars
        float val = 0.0f;
        if (dd >= 0) {
            float acc = 0.0f;
            for (int k = 0; k < 32; ++k)
                acc += Wk[(v * HIDD + dd) * 32 + k] * s_read[r * 32 + k];
            val = acc * scale;
        }
        S[j] = val;
    }
}

// ---------------------------------------------------------------------------
// Prep B: build f16 B-matrix fragments (contiguous k = 16h + s, same order as the A loads, wave32):
//   VGPR i, lane l<16  : packed (K=2i,   N=l), (K=2i+1,   N=l)
//   VGPR i, lane l>=16 : packed (K=16+2i,N=l-16),(K=17+2i,N=l-16)
// Layouts: Sfrag[ks10][i][lane], W1frag[r][ks7][n1_8][i][lane], W2frag[r][ks4][n2_7][i][lane]
__global__ __launch_bounds__(256) void prep_frags(
    const float* __restrict__ S,    // (8,300)
    const float* __restrict__ W1,   // (4,200,128)
    const float* __restrict__ W2,   // (4,128,100)
    uint32_t* __restrict__ Sfrag,
    uint32_t* __restrict__ W1frag,
    uint32_t* __restrict__ W2frag)
{
    const int NS = 10 * 8 * 32;            // 2560
    const int N1 = NRULES * 7 * 8 * 8 * 32;  // 57344
    const int N2 = NRULES * 4 * 7 * 8 * 32;  // 28672
    const int total = NS + N1 + N2;
    for (int j = blockIdx.x * 256 + threadIdx.x; j < total; j += gridDim.x * 256) {
        if (j < NS) {
            const int jj = j;
            const int lane = jj & 31, i = (jj >> 5) & 7, ks = jj >> 8;
            const int K = ks * 32 + 2 * i + ((lane & 16) ? 16 : 0);
            const int N = lane & 15;
            float lo = 0.f, hi = 0.f;
            if (N < 8) {
                if (K     < SCORK) lo = S[N * SCORK + K];
                if (K + 1 < SCORK) hi = S[N * SCORK + K + 1];
            }
            Sfrag[jj] = pack_h2(lo, hi);
        } else if (j < NS + N1) {
            const int jj = j - NS;
            const int lane = jj & 31, i = (jj >> 5) & 7;
            const int n1 = (jj >> 8) & 7;
            const int rk = jj >> 11;            // r*7 + ks
            const int ks = rk % 7, r = rk / 7;
            const int K = ks * 32 + 2 * i + ((lane & 16) ? 16 : 0);
            const int N = n1 * 16 + (lane & 15);
            float lo = 0.f, hi = 0.f;
            if (K     < HIDD) lo = W1[(r * HIDD + K) * MLPH + N];
            if (K + 1 < HIDD) hi = W1[(r * HIDD + K + 1) * MLPH + N];
            W1frag[jj] = pack_h2(lo, hi);
        } else {
            const int jj = j - NS - N1;
            const int lane = jj & 31, i = (jj >> 5) & 7;
            const int rest = jj >> 8;           // (r*4+ks)*7 + n2
            const int n2 = rest % 7;
            const int rk = rest / 7;
            const int ks = rk & 3, r = rk >> 2;
            const int K = ks * 32 + 2 * i + ((lane & 16) ? 16 : 0);
            const int N = n2 * 16 + (lane & 15);
            float lo = 0.f, hi = 0.f;
            if (N < VARD) {
                lo = W2[(r * MLPH + K) * VARD + N];
                hi = W2[(r * MLPH + K + 1) * VARD + N];
            }
            W2frag[jj] = pack_h2(lo, hi);
        }
    }
}

// ---------------------------------------------------------------------------
// Fused main kernel: scores (WMMA) -> argmax select -> gated grouped MLP (WMMA).
// 1 WG = 256 threads = 8 waves; each wave owns 16 batch rows for all 4 rules.
__global__ __launch_bounds__(256) void rule_main(
    const float* __restrict__ hidden,   // (B,3,100) flat
    const uint32_t* __restrict__ Sfrag,
    const uint32_t* __restrict__ W1frag,
    const uint32_t* __restrict__ W2frag,
    float* __restrict__ out)            // (B,100)
{
    __shared__ half_t s_h16[BATCH_TILE * HSTRIDE];                       // 82944 B
    __shared__ float  s_sc[BATCH_TILE * 8];                              //  4096 B
    __shared__ __attribute__((aligned(8))) unsigned char s_sel[BATCH_TILE]; // 128 B
    __shared__ half_t s_mid[NWAVES * 16 * MIDSTRIDE];                    // 33792 B

    const int tid  = threadIdx.x;
    const int lane = tid & 31;
    const int w    = tid >> 5;
    const int b0   = blockIdx.x * BATCH_TILE;
    const int m    = lane & 15;                 // A-matrix row owned by this lane
    const int hihalf = (lane & 16) ? 1 : 0;

    // ---- Stage hidden tile f32 -> f16 into LDS (coalesced float4 loads) ----
    {
        const float4* src = (const float4*)(hidden + (size_t)b0 * SCORK);
        const int total4 = BATCH_TILE * SCORK / 4;  // 9600
        for (int j = tid; j < total4; j += 256) {
            const float4 val = src[j];
            const int e = j * 4;
#pragma unroll
            for (int q = 0; q < 4; ++q) {
                const int ee  = e + q;
                const int row = ee / SCORK;
                const int col = ee - row * SCORK;
                s_h16[row * HSTRIDE + col] = (half_t)((&val.x)[q]);
            }
        }
        for (int j = tid; j < BATCH_TILE * 24; j += 256) {   // zero K-pad 300..323
            const int row = j / 24, col = SCORK + (j % 24);
            s_h16[row * HSTRIDE + col] = (half_t)0.0f;
        }
    }
    __syncthreads();
    // Everything below is wave-private LDS traffic (DS ops in-order per wave).

    // ---- Scores: (16 rows x 320) @ Sfrag(320 x 16, cols>=8 zero) ----
    {
        v8f acc = zero8();
        const int lr = w * 16 + m;
        const int khalf = hihalf * 16;
        for (int ks = 0; ks < 10; ++ks) {
            FragH a;
#pragma unroll
            for (int i = 0; i < 8; ++i) {
                const int koff = 2 * i + khalf;
                a.u[i] = *(const uint32_t*)&s_h16[lr * HSTRIDE + ks * 32 + koff];
            }
            FragH b;
#pragma unroll
            for (int i = 0; i < 8; ++i) b.u[i] = Sfrag[(ks * 8 + i) * 32 + lane];
            acc = __builtin_amdgcn_wmma_f32_16x16x32_f16(
                false, a.v, false, b.v, (short)0, acc, false, false);
        }
#pragma unroll
        for (int i = 0; i < 8; ++i) {                 // D: (M=i+8*hi, N=lane&15)
            const int M = i + hihalf * 8;
            const int n = lane & 15;
            if (n < 8) s_sc[(w * 16 + M) * 8 + n] = acc[i];
        }
        if (lane < 16) {                              // first-max argmax (jnp semantics)
            const int lr2 = w * 16 + lane;
            float best = s_sc[lr2 * 8];
            int   bi   = 0;
#pragma unroll
            for (int p = 1; p < 8; ++p) {
                const float v = s_sc[lr2 * 8 + p];
                if (v > best) { best = v; bi = p; }
            }
            s_sel[lr2] = (unsigned char)bi;           // p = r*2 + v
        }
    }

    // Per-lane selection of the A-row this lane feeds
    const int myrow = w * 16 + m;
    const int sv    = s_sel[myrow] & 1;               // selected variable

    // Selection bytes for the 8 C/D rows this lane touches (M = i + 8*hihalf)
    uint32_t selpack0, selpack1;
    {
        const uint32_t* ps = (const uint32_t*)&s_sel[w * 16 + hihalf * 8];
        selpack0 = ps[0]; selpack1 = ps[1];
    }

    // ---- Preload GEMM1 A fragments: rule_mlp_input = [x_{v*}, x_{1-v*}], K pad 224
    FragH a1[7];
    {
        const int khalf = hihalf * 16;
#pragma unroll
        for (int ks = 0; ks < 7; ++ks) {
#pragma unroll
            for (int i = 0; i < 8; ++i) {
                const int koff = 2 * i + khalf;
                const int k = ks * 32 + koff;
                uint32_t d = 0u;
                if (k < HIDD) {
                    const int col = (k < VARD) ? (sv * VARD + k)
                                               : ((1 - sv) * VARD + (k - VARD));
                    d = *(const uint32_t*)&s_h16[myrow * HSTRIDE + col];
                }
                a1[ks].u[i] = d;
            }
        }
    }

    v8f outacc[7];
#pragma unroll
    for (int n = 0; n < 7; ++n) outacc[n] = zero8();

    half_t* midw = &s_mid[w * 16 * MIDSTRIDE];

    for (int r = 0; r < NRULES; ++r) {
        // GEMM1: (16x224) @ W1[r](224x128), relu, rule-gate, stash f16 in LDS
#pragma unroll
        for (int n1 = 0; n1 < 8; ++n1) {
            v8f acc = zero8();
#pragma unroll
            for (int ks = 0; ks < 7; ++ks) {
                FragH b;
                const uint32_t* wp = W1frag + (((r * 7 + ks) * 8 + n1) * 8) * 32;
#pragma unroll
                for (int i = 0; i < 8; ++i) b.u[i] = wp[i * 32 + lane];
                acc = __builtin_amdgcn_wmma_f32_16x16x32_f16(
                    false, a1[ks].v, false, b.v, (short)0, acc, false, false);
            }
#pragma unroll
            for (int i = 0; i < 8; ++i) {
                const uint32_t sb = ((i < 4) ? (selpack0 >> (8 * i))
                                             : (selpack1 >> (8 * (i - 4)))) & 0xffu;
                const int rsel = (int)(sb >> 1);
                float v = acc[i];
                v = (v > 0.0f) ? v : 0.0f;            // relu
                v = (rsel == r) ? v : 0.0f;           // rule one-hot gate
                const int row = i + hihalf * 8;
                midw[row * MIDSTRIDE + n1 * 16 + (lane & 15)] = (half_t)v;
            }
        }
        // GEMM2: (16x128) @ W2[r](128x112, cols>=100 zero), accumulate over rules
        FragH a2[4];
        {
            const int khalf = hihalf * 16;
#pragma unroll
            for (int ks = 0; ks < 4; ++ks) {
#pragma unroll
                for (int i = 0; i < 8; ++i) {
                    const int koff = 2 * i + khalf;
                    a2[ks].u[i] = *(const uint32_t*)&midw[m * MIDSTRIDE + ks * 32 + koff];
                }
            }
        }
#pragma unroll
        for (int n2 = 0; n2 < 7; ++n2) {
#pragma unroll
            for (int ks = 0; ks < 4; ++ks) {
                FragH b;
                const uint32_t* wp = W2frag + (((r * 4 + ks) * 7 + n2) * 8) * 32;
#pragma unroll
                for (int i = 0; i < 8; ++i) b.u[i] = wp[i * 32 + lane];
                outacc[n2] = __builtin_amdgcn_wmma_f32_16x16x32_f16(
                    false, a2[ks].v, false, b.v, (short)0, outacc[n2], false, false);
            }
        }
    }

    // ---- Store output (B,100) f32, coalesced 16-lane rows ----
#pragma unroll
    for (int n2 = 0; n2 < 7; ++n2) {
        const int ncol = n2 * 16 + (lane & 15);
        if (ncol < VARD) {
#pragma unroll
            for (int i = 0; i < 8; ++i) {
                const int row = b0 + w * 16 + i + hihalf * 8;
                out[(size_t)row * VARD + ncol] = outacc[n2][i];
            }
        }
    }
}

// ---------------------------------------------------------------------------
extern "C" void kernel_launch(void* const* d_in, const int* in_sizes, int n_in,
                              void* d_out, int out_size, void* d_ws, size_t ws_size,
                              hipStream_t stream) {
    const float* hidden   = (const float*)d_in[0];
    const float* rule_emb = (const float*)d_in[1];
    const float* Wq       = (const float*)d_in[2];
    const float* bq       = (const float*)d_in[3];
    const float* Wk       = (const float*)d_in[4];
    const float* W1       = (const float*)d_in[5];
    const float* W2       = (const float*)d_in[6];
    float* out = (float*)d_out;

    char* ws = (char*)d_ws;
    float*    S      = (float*)(ws + WS_S);
    uint32_t* Sfrag  = (uint32_t*)(ws + WS_SFRAG);
    uint32_t* W1frag = (uint32_t*)(ws + WS_W1F);
    uint32_t* W2frag = (uint32_t*)(ws + WS_W2F);

    const int B = in_sizes[0] / 300;   // 65536

    prep_scores<<<1, 256, 0, stream>>>(rule_emb, Wq, bq, Wk, S);
    prep_frags<<<96, 256, 0, stream>>>(S, W1, W2, Sfrag, W1frag, W2frag);
    rule_main<<<B / BATCH_TILE, 256, 0, stream>>>(hidden, Sfrag, W1frag, W2frag, out);
}
